// ChebConv_Qin_Direct_64364379898355
// MI455X (gfx1250) — matrix operand order repaired and checked by reading, not compiled, not run on hardware
//
#include <hip/hip_runtime.h>
#include <hip/hip_bf16.h>
#include <math.h>

// ---------------------------------------------------------------------------
// ChebConv (complex, K=2) for MI455X / gfx1250, bf16 WMMA path.
//   A_0 = i*X ; A_1 = B@A_0 ; A_2 = 2*B@A_1 - A_0 with B = conj(L1)
//   out_re = [Re A0|Re A1|Re A2] @ Wcat + bias ; same for imag.
// Propagation GEMMs use 32x32 complex register tiles per wave:
//   16 fragment loads : 16 WMMAs per K=32 step (1 load/WMMA).
// ---------------------------------------------------------------------------

#define NN_NODES 2048
#define NUM_EDGES 65536
#define IN_C 512
#define OUT_C 512
#define KP1 3
#define KCAT (KP1 * IN_C)  // 1536

typedef __bf16 bf16_t;
typedef __attribute__((ext_vector_type(16))) __bf16 v16bf;
typedef __attribute__((ext_vector_type(8)))  float  v8f;

union FragBF { v16bf v; uint4 q[2]; };
union AccU   { v8f v; float f[8]; };
union PackU  { uint4 q; bf16_t e[8]; };

// ---------------------------------------------------------------------------
// Build kernels
// ---------------------------------------------------------------------------
__global__ void k_zero_f32(float* __restrict__ p, size_t n) {
  size_t i = (size_t)blockIdx.x * blockDim.x + threadIdx.x;
  size_t stride = (size_t)gridDim.x * blockDim.x;
  for (; i < n; i += stride) p[i] = 0.0f;
}

__global__ void k_scatter(const long long* __restrict__ edges,
                          const float* __restrict__ w,
                          float* __restrict__ A, int E, int N) {
  int t = blockIdx.x * blockDim.x + threadIdx.x;
  if (t < E) {
    int f = (int)edges[t];
    int e = (int)edges[(size_t)E + t];
    atomicAdd(&A[(size_t)f * N + e], w[t]);
  }
}

// deg_j = 0.5*(colsum_j + rowsum_j) of A  (== A_sym.sum(axis=0))
__global__ void k_deg(const float* __restrict__ A, float* __restrict__ deg, int N) {
  __shared__ float red[256];
  int j = blockIdx.x;
  float s = 0.0f;
  for (int i = threadIdx.x; i < N; i += blockDim.x)
    s += A[(size_t)i * N + j] + A[(size_t)j * N + i];
  red[threadIdx.x] = s;
  __syncthreads();
  for (int off = 128; off > 0; off >>= 1) {
    if (threadIdx.x < off) red[threadIdx.x] += red[threadIdx.x + off];
    __syncthreads();
  }
  if (threadIdx.x == 0) deg[j] = 0.5f * red[0];
}

__global__ void k_dinv(const float* __restrict__ deg, float* __restrict__ dinv, int N) {
  int j = blockIdx.x * blockDim.x + threadIdx.x;
  if (j < N) {
    float d = deg[j];
    d = (d == 0.0f) ? 1.0f : d;
    dinv[j] = rsqrtf(d);
  }
}

// B = conj(L1): B_re = -cos(theta)*A_norm ; B_im = +sin(theta)*A_norm  (bf16)
__global__ void k_build_G(const float* __restrict__ A, const float* __restrict__ dinv,
                          const float* __restrict__ qp,
                          bf16_t* __restrict__ Gre, bf16_t* __restrict__ Gim, int N) {
  const float twopi_q = 6.283185307179586f * qp[0];
  size_t total = (size_t)N * N;
  size_t idx = (size_t)blockIdx.x * blockDim.x + threadIdx.x;
  size_t stride = (size_t)gridDim.x * blockDim.x;
  for (; idx < total; idx += stride) {
    int i = (int)(idx / N);
    int j = (int)(idx % N);
    float a  = A[idx];
    float at = A[(size_t)j * N + i];
    float anorm = dinv[i] * 0.5f * (a + at) * dinv[j];
    float theta = twopi_q * (a - at);
    float s, c;
    __sincosf(theta, &s, &c);
    Gre[idx] = (bf16_t)(-c * anorm);
    Gim[idx] = (bf16_t)( s * anorm);
  }
}

// A0 = i*X: A0_re=-Xi, A0_im=Xr. Write transposed [C,N] (GEMM-B operand)
// and into Acat slice 0 (row-major [N, 3C], final-GEMM A operand).
__global__ void k_prep_x(const float* __restrict__ Xr, const float* __restrict__ Xi,
                         bf16_t* __restrict__ A0t_re, bf16_t* __restrict__ A0t_im,
                         bf16_t* __restrict__ Acat_re, bf16_t* __restrict__ Acat_im,
                         int N, int C, int acat_ld) {
  size_t total = (size_t)N * C;
  size_t idx = (size_t)blockIdx.x * blockDim.x + threadIdx.x;
  size_t stride = (size_t)gridDim.x * blockDim.x;
  for (; idx < total; idx += stride) {
    int n = (int)(idx / C);
    int c = (int)(idx % C);
    float xr = Xr[idx], xi = Xi[idx];
    bf16_t a0r = (bf16_t)(-xi);
    bf16_t a0i = (bf16_t)(xr);
    A0t_re[(size_t)c * N + n] = a0r;
    A0t_im[(size_t)c * N + n] = a0i;
    Acat_re[(size_t)n * acat_ld + c] = a0r;
    Acat_im[(size_t)n * acat_ld + c] = a0i;
  }
}

// Wt[o][k*C + f] = weight[k][f][o]  (column-major Wcat for GEMM-B operand)
__global__ void k_prep_w(const float* __restrict__ W, bf16_t* __restrict__ Wt,
                         int C, int O, int KT) {
  size_t total = (size_t)KT * C * O;
  size_t idx = (size_t)blockIdx.x * blockDim.x + threadIdx.x;
  size_t stride = (size_t)gridDim.x * blockDim.x;
  for (; idx < total; idx += stride) {
    int k = (int)(idx / ((size_t)C * O));
    int r = (int)(idx % ((size_t)C * O));
    int f = r / O;
    int o = r % O;
    Wt[(size_t)o * (KT * C) + (size_t)k * C + f] = (bf16_t)W[idx];
  }
}

// ---------------------------------------------------------------------------
// WMMA fragment helpers (bf16 16x16x32, wave32)
//  A operand: row-major src; lane m=lane&15, half=lane>>4;
//             elems 0..7 -> K = k0+half*8+j ; elems 8..15 -> K = k0+16+half*8+j
//  B operand: column-major src (rows = output columns); lane n=lane&15;
//             elems 0..7 -> K = k0+half*8+j ; elems 8..15 -> K = k0+16+half*8+j
// ---------------------------------------------------------------------------
__device__ __forceinline__ v16bf frag_a(const bf16_t* row_half_p, int k0) {
  FragBF f;
  f.q[0] = *(const uint4*)(row_half_p + k0);
  f.q[1] = *(const uint4*)(row_half_p + k0 + 16);
  return f.v;
}
__device__ __forceinline__ v16bf frag_b(const bf16_t* col_half_p, int k0) {
  FragBF f;
  f.q[0] = *(const uint4*)(col_half_p + k0);
  f.q[1] = *(const uint4*)(col_half_p + k0 + 16);
  return f.v;
}

// ---------------------------------------------------------------------------
// Complex GEMM: Y[M,Nc] = G[M,K] (complex bf16) @ H (complex, stored
// transposed [Nc,K] bf16). 32x32 complex tile per wave (2x2 sub-tiles),
// 4 waves/block along N: every fragment reused twice -> 1 load per WMMA.
// Epilogue: optional Y = 2*Y - A0 (Chebyshev step), write Acat slice and
// optionally transposed copy for the next propagation.
// ---------------------------------------------------------------------------
template <bool FUSE_CHEB2, bool WRITE_T>
__global__ __launch_bounds__(128)
void cgemm_wmma_kernel(const bf16_t* __restrict__ Gre, const bf16_t* __restrict__ Gim,
                       const bf16_t* __restrict__ Ht_re, const bf16_t* __restrict__ Ht_im,
                       bf16_t* __restrict__ Acat_re, bf16_t* __restrict__ Acat_im,
                       int acat_col0, int acat_ld,
                       bf16_t* __restrict__ Yt_re, bf16_t* __restrict__ Yt_im,
                       const bf16_t* __restrict__ A0_re, const bf16_t* __restrict__ A0_im,
                       int M, int Ncols, int K) {
  const int lane = threadIdx.x & 31;
  const int wave = threadIdx.x >> 5;
  const int colGroups = Ncols >> 7;                // block = 4 waves * 32 cols
  const int bm = blockIdx.x / colGroups;
  const int bg = blockIdx.x % colGroups;
  const int row0 = bm << 5;                        // 32 rows per wave
  const int col0 = (bg * 4 + wave) << 5;           // 32 cols per wave
  const int hv  = lane >> 4;
  const int l15 = lane & 15;

  v8f acc_rr[2][2] = {};
  v8f acc_ii[2][2] = {};
  v8f acc_ri[2][2] = {};
  v8f acc_ir[2][2] = {};

  const bf16_t* ga_p[2];
  const bf16_t* gb_p[2];
  const bf16_t* ha_p[2];
  const bf16_t* hb_p[2];
#pragma unroll
  for (int mi = 0; mi < 2; ++mi) {
    ga_p[mi] = Gre + (size_t)(row0 + mi * 16 + l15) * K + hv * 8;
    gb_p[mi] = Gim + (size_t)(row0 + mi * 16 + l15) * K + hv * 8;
  }
#pragma unroll
  for (int ni = 0; ni < 2; ++ni) {
    ha_p[ni] = Ht_re + (size_t)(col0 + ni * 16 + l15) * K + hv * 8;
    hb_p[ni] = Ht_im + (size_t)(col0 + ni * 16 + l15) * K + hv * 8;
  }

  for (int k0 = 0; k0 < K; k0 += 32) {
    if (k0 + 256 < K) {                  // global_prefetch_b8 on streamed G rows
      __builtin_prefetch(ga_p[0] + k0 + 256, 0, 1);
      __builtin_prefetch(gb_p[0] + k0 + 256, 0, 1);
    }
    v16bf ga[2], gb[2], ha[2], hb[2];
#pragma unroll
    for (int mi = 0; mi < 2; ++mi) {
      ga[mi] = frag_a(ga_p[mi], k0);
      gb[mi] = frag_a(gb_p[mi], k0);
    }
#pragma unroll
    for (int ni = 0; ni < 2; ++ni) {
      ha[ni] = frag_b(ha_p[ni], k0);
      hb[ni] = frag_b(hb_p[ni], k0);
    }
#pragma unroll
    for (int mi = 0; mi < 2; ++mi) {
#pragma unroll
      for (int ni = 0; ni < 2; ++ni) {
        acc_rr[mi][ni] = __builtin_amdgcn_wmma_f32_16x16x32_bf16(
            false, ga[mi], false, ha[ni], (short)0, acc_rr[mi][ni], false, false);
        acc_ii[mi][ni] = __builtin_amdgcn_wmma_f32_16x16x32_bf16(
            false, gb[mi], false, hb[ni], (short)0, acc_ii[mi][ni], false, false);
        acc_ri[mi][ni] = __builtin_amdgcn_wmma_f32_16x16x32_bf16(
            false, ga[mi], false, hb[ni], (short)0, acc_ri[mi][ni], false, false);
        acc_ir[mi][ni] = __builtin_amdgcn_wmma_f32_16x16x32_bf16(
            false, gb[mi], false, ha[ni], (short)0, acc_ir[mi][ni], false, false);
      }
    }
  }

#pragma unroll
  for (int mi = 0; mi < 2; ++mi) {
#pragma unroll
    for (int ni = 0; ni < 2; ++ni) {
      AccU urr, uii, uri, uir;
      urr.v = acc_rr[mi][ni];
      uii.v = acc_ii[mi][ni];
      uri.v = acc_ri[mi][ni];
      uir.v = acc_ir[mi][ni];

      const int rbase = row0 + mi * 16 + hv * 8;
      const int ncol  = col0 + ni * 16 + l15;

      float re[8], im[8];
#pragma unroll
      for (int r = 0; r < 8; ++r) {
        re[r] = urr.f[r] - uii.f[r];
        im[r] = uri.f[r] + uir.f[r];
      }

      if constexpr (FUSE_CHEB2) {
#pragma unroll
        for (int r = 0; r < 8; ++r) {
          int m = rbase + r;
          re[r] = 2.0f * re[r] - (float)A0_re[(size_t)m * acat_ld + ncol];
          im[r] = 2.0f * im[r] - (float)A0_im[(size_t)m * acat_ld + ncol];
        }
      }

#pragma unroll
      for (int r = 0; r < 8; ++r) {
        int m = rbase + r;
        Acat_re[(size_t)m * acat_ld + acat_col0 + ncol] = (bf16_t)re[r];
        Acat_im[(size_t)m * acat_ld + acat_col0 + ncol] = (bf16_t)im[r];
      }

      if constexpr (WRITE_T) {
        PackU tre, tim;
#pragma unroll
        for (int r = 0; r < 8; ++r) {
          tre.e[r] = (bf16_t)re[r];
          tim.e[r] = (bf16_t)im[r];
        }
        *(uint4*)(Yt_re + (size_t)ncol * M + rbase) = tre.q;
        *(uint4*)(Yt_im + (size_t)ncol * M + rbase) = tim.q;
      }
    }
  }
}

// ---------------------------------------------------------------------------
// Final GEMM: out_{re,im}[M,Nc] = Acat_{re,im}[M,K] @ Wcat + bias
// (Wcat stored transposed [Nc,K]; W fragment shared by both planes)
// ---------------------------------------------------------------------------
__global__ __launch_bounds__(128)
void out_gemm_kernel(const bf16_t* __restrict__ Acat_re, const bf16_t* __restrict__ Acat_im,
                     const bf16_t* __restrict__ Wt, const float* __restrict__ bias,
                     float* __restrict__ out_real, float* __restrict__ out_imag,
                     int M, int Nc, int K) {
  const int lane = threadIdx.x & 31;
  const int wave = threadIdx.x >> 5;
  const int tnGroups = Nc >> 6;
  const int bm = blockIdx.x / tnGroups;
  const int bg = blockIdx.x % tnGroups;
  const int row0 = bm << 4;
  const int col0 = (bg * 4 + wave) << 4;
  const int hv   = lane >> 4;
  const int mrow = row0 + (lane & 15);
  const int ncol = col0 + (lane & 15);

  v8f acc_re = {}, acc_im = {};

  const bf16_t* ar_p = Acat_re + (size_t)mrow * K + hv * 8;
  const bf16_t* ai_p = Acat_im + (size_t)mrow * K + hv * 8;
  const bf16_t* w_p  = Wt + (size_t)ncol * K + hv * 8;

  for (int k0 = 0; k0 < K; k0 += 32) {
    v16bf fa = frag_a(ar_p, k0);
    v16bf fb = frag_a(ai_p, k0);
    v16bf fw = frag_b(w_p, k0);
    acc_re = __builtin_amdgcn_wmma_f32_16x16x32_bf16(false, fa, false, fw, (short)0, acc_re, false, false);
    acc_im = __builtin_amdgcn_wmma_f32_16x16x32_bf16(false, fb, false, fw, (short)0, acc_im, false, false);
  }

  AccU ur, ui;
  ur.v = acc_re; ui.v = acc_im;
  float b = bias[ncol];
#pragma unroll
  for (int r = 0; r < 8; ++r) {
    int m = row0 + hv * 8 + r;
    out_real[(size_t)m * Nc + ncol] = ur.f[r] + b;
    out_imag[(size_t)m * Nc + ncol] = ui.f[r] + b;
  }
}

// ---------------------------------------------------------------------------
// Launch
// ---------------------------------------------------------------------------
extern "C" void kernel_launch(void* const* d_in, const int* in_sizes, int n_in,
                              void* d_out, int out_size, void* d_ws, size_t ws_size,
                              hipStream_t stream) {
  (void)n_in; (void)out_size; (void)ws_size;
  const int N = NN_NODES, E = NUM_EDGES;

  const float*     Xr    = (const float*)d_in[0];
  const float*     Xi    = (const float*)d_in[1];
  const long long* edges = (const long long*)d_in[2];
  const float*     qp    = (const float*)d_in[3];
  const float*     ew    = (const float*)d_in[4];
  const float*     W     = (const float*)d_in[5];
  const float*     bias  = (const float*)d_in[6];
  (void)in_sizes;

  float* out_real = (float*)d_out;
  float* out_imag = out_real + (size_t)N * OUT_C;

  // Workspace carve-up (all offsets 16B-aligned)
  char* ws = (char*)d_ws;
  float*  A_dense = (float*)(ws);                               // 16 MB
  bf16_t* Gre     = (bf16_t*)(ws + 16777216);                   //  8 MB
  bf16_t* Gim     = (bf16_t*)(ws + 25165824);                   //  8 MB
  bf16_t* A0t_re  = (bf16_t*)(ws + 33554432);                   //  2 MB
  bf16_t* A0t_im  = (bf16_t*)(ws + 35651584);                   //  2 MB
  bf16_t* A1t_re  = (bf16_t*)(ws + 37748736);                   //  2 MB
  bf16_t* A1t_im  = (bf16_t*)(ws + 39845888);                   //  2 MB
  bf16_t* Acat_re = (bf16_t*)(ws + 41943040);                   //  6 MB
  bf16_t* Acat_im = (bf16_t*)(ws + 48234496);                   //  6 MB
  bf16_t* Wt      = (bf16_t*)(ws + 54525952);                   // 1.5 MB
  float*  deg     = (float*)(ws + 56098816);
  float*  dinv    = (float*)(ws + 56107008);

  // 1) dense adjacency
  k_zero_f32<<<16384, 256, 0, stream>>>(A_dense, (size_t)N * N);
  k_scatter<<<(E + 255) / 256, 256, 0, stream>>>(edges, ew, A_dense, E, N);

  // 2) degree / D^-1/2
  k_deg<<<N, 256, 0, stream>>>(A_dense, deg, N);
  k_dinv<<<(N + 255) / 256, 256, 0, stream>>>(deg, dinv, N);

  // 3) B = conj(L1) in bf16
  k_build_G<<<16384, 256, 0, stream>>>(A_dense, dinv, qp, Gre, Gim, N);

  // 4) operand prep (bf16, transposed where needed)
  k_prep_x<<<4096, 256, 0, stream>>>(Xr, Xi, A0t_re, A0t_im, Acat_re, Acat_im,
                                     N, IN_C, KCAT);
  k_prep_w<<<3072, 256, 0, stream>>>(W, Wt, IN_C, OUT_C, KP1);

  // 5) A1 = B @ A0  (write Acat slice 1 + transposed copy)
  cgemm_wmma_kernel<false, true><<<(N / 32) * (IN_C / 128), 128, 0, stream>>>(
      Gre, Gim, A0t_re, A0t_im, Acat_re, Acat_im, /*acat_col0=*/IN_C, KCAT,
      A1t_re, A1t_im, nullptr, nullptr, N, IN_C, N);

  // 6) A2 = 2*B @ A1 - A0  (write Acat slice 2)
  cgemm_wmma_kernel<true, false><<<(N / 32) * (IN_C / 128), 128, 0, stream>>>(
      Gre, Gim, A1t_re, A1t_im, Acat_re, Acat_im, /*acat_col0=*/2 * IN_C, KCAT,
      nullptr, nullptr, Acat_re, Acat_im, N, IN_C, N);

  // 7) out = Acat @ Wcat + bias (both planes, shared W fragments)
  out_gemm_kernel<<<(N / 16) * (OUT_C / 64), 128, 0, stream>>>(
      Acat_re, Acat_im, Wt, bias, out_real, out_imag, N, OUT_C, KCAT);
}
